// DeformConvV3_50912542326882
// MI455X (gfx1250) — matrix operand order repaired and checked by reading, not compiled, not run on hardware
//
#include <hip/hip_runtime.h>
#include <hip/hip_bf16.h>
#include <math.h>

typedef __attribute__((ext_vector_type(2)))  float    v2f;
typedef __attribute__((ext_vector_type(4)))  float    v4f;
typedef __attribute__((ext_vector_type(8)))  float    v8f;
typedef __attribute__((ext_vector_type(4)))  unsigned v4u;
typedef __attribute__((ext_vector_type(16))) __bf16   v16bf;

#define C1     128
#define C2     128
#define HIMG   80
#define WIMG   80
#define PIX    6400      // 80*80
#define BATCH  8
#define KTAPS  9
#define KDIM   1152      // C1*KTAPS
#define OFFC   18
#define MASKC  9

// ---- workspace layout (dword offsets), total ~32.3 MB ----
#define XNHWC_OFF  0u
#define OFFW_OFF   6553600u           // [B*PIX][18]
#define MASKW_OFF  7475200u           // [B*PIX][9]
#define WDEF_OFF   7936000u           // bf16 B panels: 36 kchunks * 8 nt * 32 lanes * 8 dw
#define WOFFM_OFF  8009728u           // f32 B panels: 288 * 2 * 32 * 2
#define W1_OFF     8046592u           // f32 B panels: 40 * 8 * 32 * 2

union BFVec { v4u q[2]; v16bf v; };

__device__ __forceinline__ unsigned pack_bf16x2(float lo, float hi) {
  unsigned a = __float_as_uint(lo);
  unsigned b = __float_as_uint(hi);
  a += 0x7FFFu + ((a >> 16) & 1u);   // round-to-nearest-even
  b += 0x7FFFu + ((b >> 16) & 1u);
  return (a >> 16) | (b & 0xFFFF0000u);
}

// ---------------- NCHW -> NHWC, LDS tile transpose (coalesced both sides) ----------------
__global__ void __launch_bounds__(256) k_transpose(const float* __restrict__ x,
                                                   float* __restrict__ xn) {
  __shared__ float tile[32][65];
  int blk = blockIdx.x;                // b*400 + cb*100 + pb
  int b   = blk / 400;
  int r   = blk - b * 400;
  int cb  = r / 100;
  int pb  = r - cb * 100;
  int c0  = cb * 32, p0 = pb * 64;
  int t   = threadIdx.x;

  int tx = t & 63, ty = t >> 6;        // read: 64 consecutive pixels per row
  for (int cc = 0; cc < 8; ++cc) {
    int c = cc * 4 + ty;
    tile[c][tx] = x[((size_t)(b * C1 + c0 + c)) * PIX + p0 + tx];
  }
  __syncthreads();
  int cx = t & 31, py = t >> 5;        // write: 32 consecutive channels per pixel
  for (int pp = 0; pp < 8; ++pp) {
    int p = pp * 8 + py;
    xn[((size_t)b * PIX + p0 + p) * C1 + c0 + cx] = tile[cx][p];
  }
}

// ---------------- pack w_def into bf16 WMMA-B panels ----------------
// B[k][n] with k = tap*128 + c, element = w_def[n*1152 + c*9 + tap]
__global__ void k_pack_wdef(const float* __restrict__ wdef, unsigned* __restrict__ wp) {
  int blk = blockIdx.x;                // kc*8 + nt, kc in [0,36)
  int kc  = blk >> 3;
  int l   = threadIdx.x;
  int klo = (l < 16) ? 0 : 16;
  int n   = ((blk & 7) * 16) + (l & 15);
  unsigned* dst = wp + ((size_t)blk * 32 + l) * 8;
  for (int d = 0; d < 8; ++d) {
    int k   = kc * 32 + klo + 2 * d;
    int tap = k / C1, c = k - tap * C1;
    float lo = wdef[(size_t)n * KDIM + c * KTAPS + tap];
    float hi = wdef[(size_t)n * KDIM + (c + 1) * KTAPS + tap];
    dst[d] = pack_bf16x2(lo, hi);
  }
}

// ---------------- pack w_off/w_mask into f32 WMMA-B panels (N: 0..17 off, 18..26 mask) ----
__global__ void k_pack_woffmask(const float* __restrict__ woff, const float* __restrict__ wmask,
                                float* __restrict__ wp) {
  int blk = blockIdx.x;                // kc*2 + nt, kc in [0,288)
  int kc  = blk >> 1;
  int l   = threadIdx.x;
  int klo = (l < 16) ? 0 : 2;
  int n   = ((blk & 1) * 16) + (l & 15);
  float* dst = wp + ((size_t)blk * 32 + l) * 2;
  for (int d = 0; d < 2; ++d) {
    int k   = kc * 4 + klo + d;
    int tap = k / C1, c = k - tap * C1;
    float v = 0.f;
    if (n < OFFC)              v = woff[(size_t)n * KDIM + c * KTAPS + tap];
    else if (n < OFFC + MASKC) v = wmask[(size_t)(n - OFFC) * KDIM + c * KTAPS + tap];
    dst[d] = v;
  }
}

// ---------------- pack w_1x1 (K=160 padded from 155) ----------------
__global__ void k_pack_w1(const float* __restrict__ w1, float* __restrict__ wp) {
  int blk = blockIdx.x;                // kc*8 + nt, kc in [0,40)
  int kc  = blk >> 3;
  int l   = threadIdx.x;
  int klo = (l < 16) ? 0 : 2;
  int n   = ((blk & 7) * 16) + (l & 15);
  float* dst = wp + ((size_t)blk * 32 + l) * 2;
  for (int d = 0; d < 2; ++d) {
    int k = kc * 4 + klo + d;
    dst[d] = (k < 155) ? w1[(size_t)n * 155 + k] : 0.f;
  }
}

// ---------------- offset/mask conv: fp32 WMMA implicit GEMM, 1 wave / 16-pixel tile ----
__global__ void __launch_bounds__(32) k_offmask(
    const float* __restrict__ xn, const float* __restrict__ wp,
    const float* __restrict__ boff, const float* __restrict__ bmask,
    float* __restrict__ offw, float* __restrict__ maskw) {
  int blk  = blockIdx.x;
  int b    = blk / (PIX / 16);
  int p0   = (blk - b * (PIX / 16)) * 16;
  int l    = threadIdx.x;
  int p    = l & 15;
  int pix  = p0 + p;
  int oy   = pix / WIMG, ox = pix - oy * WIMG;
  int kloA = (l < 16) ? 0 : 2;

  v8f acc0 = {0.f,0.f,0.f,0.f,0.f,0.f,0.f,0.f};
  v8f acc1 = acc0;
  const float* bptr = wp + l * 2;

  for (int tap = 0; tap < KTAPS; ++tap) {
    int ky = tap / 3, kx = tap - ky * 3;
    int iy = oy - 1 + ky, ix = ox - 1 + kx;
    bool valid = (iy >= 0) && (iy < HIMG) && (ix >= 0) && (ix < WIMG);
    float vmask = valid ? 1.f : 0.f;
    int iyc = min(max(iy, 0), HIMG - 1);
    int ixc = min(max(ix, 0), WIMG - 1);
    const float* row = xn + ((size_t)b * PIX + iyc * WIMG + ixc) * C1 + kloA;
    for (int kc = 0; kc < 32; ++kc) {
      v2f a = *(const v2f*)(row + kc * 4);
      a = a * vmask;                             // branch-free boundary zeroing
      int kcG = tap * 32 + kc;
      v2f b0 = *(const v2f*)(bptr + kcG * 128);
      v2f b1 = *(const v2f*)(bptr + kcG * 128 + 64);
      acc0 = __builtin_amdgcn_wmma_f32_16x16x4_f32(false, a, false, b0, (short)0, acc0, false, false);
      acc1 = __builtin_amdgcn_wmma_f32_16x16x4_f32(false, a, false, b1, (short)0, acc1, false, false);
    }
  }

  int n   = l & 15;
  int rhi = (l >> 4) * 8;
  for (int nt = 0; nt < 2; ++nt) {
    int ch = nt * 16 + n;
    v8f acc = nt ? acc1 : acc0;
    if (ch < OFFC) {
      float bias = boff[ch];
      for (int r = 0; r < 8; ++r)
        offw[((size_t)b * PIX + p0 + rhi + r) * OFFC + ch] = acc[r] + bias;
    } else if (ch < OFFC + MASKC) {
      int cm = ch - OFFC;
      float bias = bmask[cm];
      for (int r = 0; r < 8; ++r) {
        float v = acc[r] + bias;
        maskw[((size_t)b * PIX + p0 + rhi + r) * MASKC + cm] = 1.f / (1.f + __expf(-v));
      }
    }
  }
}

// ---------------- deformable conv (bf16 WMMA) + fused 1x1 (f32 WMMA) ----------------
__global__ void __launch_bounds__(128) k_deform(
    const float* __restrict__ xn, const unsigned* __restrict__ wdp,
    const float* __restrict__ w1p,
    const float* __restrict__ offw, const float* __restrict__ maskw,
    const float* __restrict__ bdef, const float* __restrict__ b1,
    float* __restrict__ out) {
  __shared__ __align__(16) unsigned short As[16][136];   // bf16 sampled panel, +8 pad
  __shared__ __align__(16) float A1[16][168];            // fp32 concat / output panel, +8 pad

  int blk  = blockIdx.x;
  int b    = blk / (PIX / 16);
  int p0   = (blk - b * (PIX / 16)) * 16;
  int t    = threadIdx.x;
  int wid  = t >> 5;
  int l    = t & 31;

  // staging mapping: thread -> (pixel, 16-channel group)
  int sp   = t >> 3;
  int cg   = (t & 7) * 16;
  int spix = p0 + sp;
  int soy  = spix / WIMG, sox = spix - soy * WIMG;

  // wmma lane mapping
  int pa   = l & 15;
  int kloA = (l < 16) ? 0 : 16;   // bf16 A frag
  int klo4 = (l < 16) ? 0 : 2;    // f32 A frag

  v8f accd0 = {0.f,0.f,0.f,0.f,0.f,0.f,0.f,0.f};
  v8f accd1 = accd0;

  for (int tap = 0; tap < KTAPS; ++tap) {
    // ---- bilinear sample + mask -> bf16 LDS panel (branch-free) ----
    float offy = offw[((size_t)b * PIX + spix) * OFFC + 2 * tap];
    float offx = offw[((size_t)b * PIX + spix) * OFFC + 2 * tap + 1];
    float m    = maskw[((size_t)b * PIX + spix) * MASKC + tap];
    int ky = tap / 3, kx = tap - ky * 3;
    float py = (float)(soy - 1 + ky) + offy;
    float px = (float)(sox - 1 + kx) + offx;
    float fy0 = floorf(py), fx0 = floorf(px);
    float wy1 = py - fy0, wx1 = px - fx0;
    float wy0 = 1.f - wy1, wx0 = 1.f - wx1;
    int y0 = (int)fy0, x0 = (int)fx0, y1 = y0 + 1, x1 = x0 + 1;
    bool vy0 = (y0 >= 0) && (y0 < HIMG), vy1 = (y1 >= 0) && (y1 < HIMG);
    bool vx0 = (x0 >= 0) && (x0 < WIMG), vx1 = (x1 >= 0) && (x1 < WIMG);
    float w00 = (vy0 && vx0) ? wy0 * wx0 * m : 0.f;
    float w01 = (vy0 && vx1) ? wy0 * wx1 * m : 0.f;
    float w10 = (vy1 && vx0) ? wy1 * wx0 * m : 0.f;
    float w11 = (vy1 && vx1) ? wy1 * wx1 * m : 0.f;
    int y0c = min(max(y0, 0), HIMG - 1), y1c = min(max(y1, 0), HIMG - 1);
    int x0c = min(max(x0, 0), WIMG - 1), x1c = min(max(x1, 0), WIMG - 1);
    const float* r00 = xn + ((size_t)b * PIX + y0c * WIMG + x0c) * C1;
    const float* r01 = xn + ((size_t)b * PIX + y0c * WIMG + x1c) * C1;
    const float* r10 = xn + ((size_t)b * PIX + y1c * WIMG + x0c) * C1;
    const float* r11 = xn + ((size_t)b * PIX + y1c * WIMG + x1c) * C1;

    for (int j = 0; j < 4; ++j) {
      int c = cg + 4 * j;
      v4f a0 = *(const v4f*)(r00 + c);
      v4f a1 = *(const v4f*)(r01 + c);
      v4f a2 = *(const v4f*)(r10 + c);
      v4f a3 = *(const v4f*)(r11 + c);
      v4f s  = a0 * w00 + a1 * w01 + a2 * w10 + a3 * w11;
      unsigned* dst = (unsigned*)&As[sp][c];
      dst[0] = pack_bf16x2(s.x, s.y);
      dst[1] = pack_bf16x2(s.z, s.w);
    }
    __syncthreads();

    // ---- bf16 WMMA over this tap's 128-channel K slab ----
    if (tap < 8)
      __builtin_prefetch(wdp + (((size_t)((tap + 1) * 4) * 8 + wid * 2) * 32 + l) * 8, 0, 1);
    for (int kc4 = 0; kc4 < 4; ++kc4) {
      BFVec a;
      a.q[0] = *(const v4u*)&As[pa][kc4 * 32 + kloA];
      a.q[1] = *(const v4u*)&As[pa][kc4 * 32 + kloA + 8];
      int kcG = tap * 4 + kc4;
      {
        BFVec bm;
        const unsigned* bp = wdp + (((size_t)kcG * 8 + wid * 2 + 0) * 32 + l) * 8;
        bm.q[0] = *(const v4u*)bp;
        bm.q[1] = *(const v4u*)(bp + 4);
        accd0 = __builtin_amdgcn_wmma_f32_16x16x32_bf16(false, a.v, false, bm.v, (short)0, accd0, false, false);
      }
      {
        BFVec bm;
        const unsigned* bp = wdp + (((size_t)kcG * 8 + wid * 2 + 1) * 32 + l) * 8;
        bm.q[0] = *(const v4u*)bp;
        bm.q[1] = *(const v4u*)(bp + 4);
        accd1 = __builtin_amdgcn_wmma_f32_16x16x32_bf16(false, a.v, false, bm.v, (short)0, accd1, false, false);
      }
    }
    __syncthreads();
  }

  // ---- build fp32 concat panel A1 = [y+b_def | offset | mask | 0] ----
  {
    int n   = l & 15;
    int rhi = (l >> 4) * 8;
    for (int nt = 0; nt < 2; ++nt) {
      int ch = wid * 32 + nt * 16 + n;
      float bias = bdef[ch];
      v8f acc = nt ? accd1 : accd0;
      for (int r = 0; r < 8; ++r)
        A1[rhi + r][ch] = acc[r] + bias;
    }
    int q = t & 7;
    for (int jj = 0; jj < 4; ++jj) {
      int k = 128 + q * 4 + jj;     // 128..159
      float v = 0.f;
      if (k < 146)      v = offw[((size_t)b * PIX + spix) * OFFC + (k - 128)];
      else if (k < 155) v = maskw[((size_t)b * PIX + spix) * MASKC + (k - 146)];
      A1[sp][k] = v;
    }
  }
  __syncthreads();

  // ---- fused 1x1 conv: f32 WMMA over K=160 ----
  v8f acco0 = {0.f,0.f,0.f,0.f,0.f,0.f,0.f,0.f};
  v8f acco1 = acco0;
  for (int kc = 0; kc < 40; ++kc) {
    v2f a = *(const v2f*)&A1[pa][kc * 4 + klo4];
    v2f bf0 = *(const v2f*)(w1p + (((size_t)kc * 8 + wid * 2 + 0) * 32 + l) * 2);
    v2f bf1 = *(const v2f*)(w1p + (((size_t)kc * 8 + wid * 2 + 1) * 32 + l) * 2);
    acco0 = __builtin_amdgcn_wmma_f32_16x16x4_f32(false, a, false, bf0, (short)0, acco0, false, false);
    acco1 = __builtin_amdgcn_wmma_f32_16x16x4_f32(false, a, false, bf1, (short)0, acco1, false, false);
  }
  __syncthreads();                  // all A1 reads complete before reuse

  // ---- epilogue: bias + transpose through LDS, coalesced NCHW store ----
  {
    int n   = l & 15;
    int rhi = (l >> 4) * 8;
    for (int nt = 0; nt < 2; ++nt) {
      int o = wid * 32 + nt * 16 + n;
      float bias = b1[o];
      v8f acc = nt ? acco1 : acco0;
      for (int r = 0; r < 8; ++r)
        A1[rhi + r][o] = acc[r] + bias;
    }
  }
  __syncthreads();
  {
    int px = t & 15, og = t >> 4;   // 8 channel groups x 16 pixels
    for (int s = 0; s < 16; ++s) {
      int o = og * 16 + s;
      out[((size_t)b * C2 + o) * PIX + p0 + px] = A1[px][o];
    }
  }
}

// ---------------- host launcher ----------------
extern "C" void kernel_launch(void* const* d_in, const int* in_sizes, int n_in,
                              void* d_out, int out_size, void* d_ws, size_t ws_size,
                              hipStream_t stream) {
  const float* x      = (const float*)d_in[0];
  const float* w_off  = (const float*)d_in[1];
  const float* b_off  = (const float*)d_in[2];
  const float* w_mask = (const float*)d_in[3];
  const float* b_mask = (const float*)d_in[4];
  const float* w_def  = (const float*)d_in[5];
  const float* b_def  = (const float*)d_in[6];
  const float* w_1x1  = (const float*)d_in[7];
  const float* b_1x1  = (const float*)d_in[8];
  float* out = (float*)d_out;

  float*    wsf   = (float*)d_ws;
  float*    xn    = wsf + XNHWC_OFF;
  float*    offw  = wsf + OFFW_OFF;
  float*    maskw = wsf + MASKW_OFF;
  unsigned* wdp   = (unsigned*)(wsf + WDEF_OFF);
  float*    woffp = wsf + WOFFM_OFF;
  float*    w1p   = wsf + W1_OFF;

  k_transpose    <<<BATCH * 400, 256, 0, stream>>>(x, xn);
  k_pack_wdef    <<<36 * 8,  32, 0, stream>>>(w_def, wdp);
  k_pack_woffmask<<<288 * 2, 32, 0, stream>>>(w_off, w_mask, woffp);
  k_pack_w1      <<<40 * 8,  32, 0, stream>>>(w_1x1, w1p);
  k_offmask      <<<BATCH * (PIX / 16), 32, 0, stream>>>(xn, woffp, b_off, b_mask, offw, maskw);
  k_deform       <<<BATCH * (PIX / 16), 128, 0, stream>>>(xn, wdp, w1p, offw, maskw, b_def, b_1x1, out);
}
